// ResidualVQ_14989435863665
// MI455X (gfx1250) — matrix operand order repaired and checked by reading, not compiled, not run on hardware
//
#include <hip/hip_runtime.h>
#include <hip/hip_bf16.h>

// ---------------- problem constants (from reference setup_inputs) -----------
#define B_   8
#define DIN  512
#define T_   4096
#define NCB  8
#define K_   2048
#define D_   128
#define TT   128      // tokens per block tile (8 waves x 16 token columns)
#define ZSTP 72       // padded bf16 row stride for the z staging buffer
#define KC   64       // codebook rows per TDM-staged LDS chunk (16 KB)

// ---- LDS partition (bytes), dynamic shared memory assumed base-0 ----------
#define RF_OFF   0
#define PJ_OFF   (TT * D_ * 4)                    // 65536
#define RTB_OFF  (2 * TT * D_ * 4)                // 131072
#define CBA0_OFF (RTB_OFF + TT * D_ * 2)          // 163840
#define CBA1_OFF (CBA0_OFF + KC * D_ * 2)         // 180224
#define AUX_OFF  (CBA1_OFF + KC * D_ * 2)         // 196608
#define AUXB     ((TT * ZSTP * 2 > K_ * 4) ? TT * ZSTP * 2 : K_ * 4)  // 18432
#define WIN_OFF  (AUX_OFF + AUXB)
#define RED_OFF  (WIN_OFF + TT * 4)
#define SMEM_TOT (RED_OFF + 256 * 4)              // ~211.5 KB

typedef __attribute__((ext_vector_type(16))) __bf16 v16bf;
typedef __attribute__((ext_vector_type(8)))  float  v8f;
typedef __attribute__((ext_vector_type(4)))  unsigned int u32x4;
typedef __attribute__((ext_vector_type(8)))  int    i32x8;
typedef __attribute__((ext_vector_type(4)))  int    i32x4;

struct Pack32 { uint4 lo, hi; };

static __device__ __forceinline__ v16bf pack_frag(uint4 lo, uint4 hi) {
  Pack32 p{lo, hi};
  return __builtin_bit_cast(v16bf, p);
}

// f32 -> bf16 round-to-nearest-even
static __device__ __forceinline__ unsigned short f2bf(float f) {
  unsigned u = __float_as_uint(f);
  u += 0x7FFFu + ((u >> 16) & 1u);
  return (unsigned short)(u >> 16);
}

// ---- Tensor Data Mover: 1-D contiguous copy global -> LDS -----------------
// D# per CDNA5 ISA: group0 = {count, lds_addr,
// global_addr, type=2}; group1 = {data_size=3 (8B), tensor_dim0 = tile_dim0 =
// nQwords, stride = nQwords}. Groups 2/3 zero (<=2-D tensor).
static __device__ __forceinline__ void tdm_copy_to_lds(
    const void* gsrc, unsigned ldsByteOff, unsigned nQwords) {
  unsigned long long ga = (unsigned long long)(size_t)gsrc;
  u32x4 g0 = { 1u,                                   // count=1 valid descriptor
               ldsByteOff,                           // LDS byte address
               (unsigned)(ga & 0xFFFFFFFFu),         // global_addr[31:0]
               (unsigned)((ga >> 32) & 0x1FFFFFFu) | (2u << 30) }; // type=2
  i32x8 g1 = { (int)(3u << 16),                      // data_size=8B, mask=0
               (int)((nQwords & 0xFFFFu) << 16),     // tensor_dim0[15:0]
               (int)((nQwords >> 16) & 0xFFFFu),     // tensor_dim0[31:16]
               (int)((nQwords & 0xFFFFu) << 16),     // tile_dim0
               0,                                    // tile_dim1/2 unused
               (int)nQwords,                         // tensor_dim0_stride lo
               0, 0 };
  i32x4 z4 = { 0, 0, 0, 0 };
#if __clang_major__ >= 23
  i32x8 z8 = { 0, 0, 0, 0, 0, 0, 0, 0 };
  __builtin_amdgcn_tensor_load_to_lds(g0, g1, z4, z4, z8, 0);
#else
  __builtin_amdgcn_tensor_load_to_lds(g0, g1, z4, z4, 0);
#endif
}

// ---------------- prep kernels ---------------------------------------------
__global__ __launch_bounds__(128) void rvq_prep_cb(
    const float* __restrict__ cb, unsigned short* __restrict__ cbh,
    float* __restrict__ cn) {
  __shared__ float partial[4];
  const int row = blockIdx.x;          // n*K rows
  const int t = threadIdx.x;           // 0..127 == D
  float v = cb[(size_t)row * D_ + t];
  cbh[(size_t)row * D_ + t] = f2bf(v);
  float sq = v * v;
  for (int off = 16; off > 0; off >>= 1) sq += __shfl_down(sq, off);
  if ((t & 31) == 0) partial[t >> 5] = sq;
  __syncthreads();
  if (t == 0) cn[row] = 0.5f * (partial[0] + partial[1] + partial[2] + partial[3]);
}

__global__ __launch_bounds__(256) void rvq_prep_w(
    const float* __restrict__ w_in, const float* __restrict__ w_out,
    unsigned short* __restrict__ wih, unsigned short* __restrict__ woh) {
  int i = blockIdx.x * blockDim.x + threadIdx.x;
  if (i < D_ * DIN) wih[i] = f2bf(w_in[i]);
  int j = i - D_ * DIN;
  if (j >= 0 && j < DIN * D_) woh[j] = f2bf(w_out[j]);
}

__global__ void rvq_zero_commit(float* p) {
  if (threadIdx.x == 0 && blockIdx.x == 0) *p = 0.0f;
}

// ---------------- fused RVQ kernel -----------------------------------------
__global__ __launch_bounds__(256) void rvq_fused(
    const float* __restrict__ z,                 // [B][Din][T]
    const unsigned short* __restrict__ w_in_h,   // [D][Din] bf16
    const float* __restrict__ b_in,              // [D]
    const unsigned short* __restrict__ w_out_h,  // [Din][D] bf16
    const float* __restrict__ b_out,             // [Din]
    const float* __restrict__ cb_f32,            // [n][K][D]
    const unsigned short* __restrict__ cb_h,     // [n][K][D] bf16
    const float* __restrict__ cnorm,             // [n][K]  (0.5*|c|^2)
    float* __restrict__ out_codes,               // [B][n][T] (as float)
    float* __restrict__ out_y,                   // [B][Din][T]
    float* __restrict__ out_commit)              // scalar
{
  extern __shared__ char smem[];
  float*          Rf   = (float*)(smem + RF_OFF);            // [TT][D] residual
  float*          Pj   = (float*)(smem + PJ_OFF);            // [TT][D] initial proj
  unsigned short* Rtb  = (unsigned short*)(smem + RTB_OFF);  // [TT][D] bf16 mirror
  unsigned short* cbA0 = (unsigned short*)(smem + CBA0_OFF); // [KC][D] stage 0
  unsigned short* cbA1 = (unsigned short*)(smem + CBA1_OFF); // [KC][D] stage 1
  char* aux = smem + AUX_OFF;
  unsigned short* zst = (unsigned short*)aux;   // [TT][ZSTP] z staging (bf16)
  float*          cnl = (float*)aux;            // aliased: [K] staged half-norms
  int*   winners = (int*)(smem + WIN_OFF);      // [TT]
  float* red     = (float*)(smem + RED_OFF);    // [256]

  const int tid  = threadIdx.x;
  const int wave = tid >> 5;
  const int lane = tid & 31;
  const int lrow = lane & 15;
  const int hf   = lane >> 4;        // half-wave select (WMMA fragment layout)
  const int tile = blockIdx.x;       // 0..T/TT-1
  const int b    = blockIdx.y;       // 0..B-1
  const int t0   = tile * TT;
  const int tokbase = wave * 16;
  const int tok  = tokbase + lrow;   // this lane's token column

  // LDS byte offsets for the TDM destination (generic ptr low 32 bits)
  const unsigned cbOff0 = (unsigned)(size_t)(void*)cbA0;
  const unsigned cbOff1 = (unsigned)(size_t)(void*)cbA1;

  // ---- Stage 1: proj = w_in @ z_tile + b_in  (M=D=128, N=TT=128, K=Din=512)
  v8f acc[8];
  const v8f vzero = {0, 0, 0, 0, 0, 0, 0, 0};
#pragma unroll
  for (int mt = 0; mt < 8; ++mt) acc[mt] = vzero;

  for (int c = 0; c < DIN / 64; ++c) {
    const float* zb = z + ((size_t)b * DIN + (size_t)c * 64) * T_ + t0;
    for (int i = 0; i < (64 * TT) / 256; ++i) {
      int idx = i * 256 + tid;
      int tt = idx & (TT - 1);
      int dl = idx >> 7;
      zst[tt * ZSTP + dl] = f2bf(zb[(size_t)dl * T_ + tt]);
    }
    __syncthreads();
#pragma unroll
    for (int ds = 0; ds < 2; ++ds) {
      const int dloc = ds * 32;
      const uint4* bp = (const uint4*)(zst + tok * ZSTP + dloc + hf * 16);
      const v16bf bf = pack_frag(bp[0], bp[1]);
#pragma unroll
      for (int mt = 0; mt < 8; ++mt) {
        const unsigned short* ap =
            w_in_h + (size_t)(mt * 16 + lrow) * DIN + c * 64 + dloc + hf * 16;
        const v16bf af = pack_frag(*(const uint4*)ap, *(const uint4*)(ap + 8));
        acc[mt] = __builtin_amdgcn_wmma_f32_16x16x32_bf16(
            false, af, false, bf, (short)0, acc[mt], false, false);
      }
    }
    __syncthreads();
  }
#pragma unroll
  for (int mt = 0; mt < 8; ++mt) {
#pragma unroll
    for (int r = 0; r < 8; ++r) {
      const int m = mt * 16 + hf * 8 + r;            // C-layout: row = r + 8*hf
      const float v = acc[mt][r] + b_in[m];
      Rf[tok * D_ + m]  = v;
      Pj[tok * D_ + m]  = v;
      Rtb[tok * D_ + m] = f2bf(v);
    }
  }
  __syncthreads();

  // ---- Stage 2: 8 RVQ steps. score[k,t] = c_k . r_t - 0.5|c_k|^2, argmax_k.
  float commitAcc = 0.0f;
  for (int s = 0; s < NCB; ++s) {
    for (int i = 0; i < K_ / 256; ++i)
      cnl[i * 256 + tid] = cnorm[(size_t)s * K_ + i * 256 + tid];

    // residual B-fragments are constant for the whole scoring sweep: hoist
    v16bf bfrag[4];
#pragma unroll
    for (int dsp = 0; dsp < 4; ++dsp) {
      const uint4* bp = (const uint4*)(Rtb + tok * D_ + dsp * 32 + hf * 16);
      bfrag[dsp] = pack_frag(bp[0], bp[1]);
    }

    const unsigned short* cbs = cb_h + (size_t)s * K_ * D_;
    // prologue: TDM chunk 0 into buffer 0 (wave 0 drives the DMA)
    if (wave == 0) {
      tdm_copy_to_lds(cbs, cbOff0, KC * D_ / 4);
      __builtin_amdgcn_s_wait_tensorcnt(0);
    }
    __syncthreads();

    float bestv = -3.4e38f;
    int   besti = 0;
    for (int c = 0; c < K_ / KC; ++c) {
      // kick next chunk's DMA while everyone computes on the current one
      if (wave == 0 && (c + 1) < K_ / KC)
        tdm_copy_to_lds(cbs + (size_t)(c + 1) * KC * D_,
                        ((c + 1) & 1) ? cbOff1 : cbOff0, KC * D_ / 4);
      const unsigned short* Ab = (c & 1) ? cbA1 : cbA0;
#pragma unroll
      for (int kbl = 0; kbl < KC / 16; ++kbl) {
        v8f accs = vzero;
        const unsigned short* arow = Ab + (kbl * 16 + lrow) * D_ + hf * 16;
#pragma unroll
        for (int dsp = 0; dsp < 4; ++dsp) {
          const v16bf af = pack_frag(*(const uint4*)(arow + dsp * 32),
                                     *(const uint4*)(arow + dsp * 32 + 8));
          accs = __builtin_amdgcn_wmma_f32_16x16x32_bf16(
              false, af, false, bfrag[dsp], (short)0, accs, false, false);
        }
#pragma unroll
        for (int r = 0; r < 8; ++r) {
          const int k = c * KC + kbl * 16 + hf * 8 + r;
          const float sc = accs[r] - cnl[k];
          if (sc > bestv) { bestv = sc; besti = k; }
        }
      }
      if (wave == 0) __builtin_amdgcn_s_wait_tensorcnt(0);
      __syncthreads();
    }
    // lanes l and l^16 hold the same token column; merge their halves
    const float ov = __shfl_xor(bestv, 16);
    const int   oi = __shfl_xor(besti, 16);
    if (ov > bestv || (ov == bestv && oi < besti)) { bestv = ov; besti = oi; }
    if (hf == 0) {
      winners[tok] = besti;
      out_codes[(size_t)b * NCB * T_ + (size_t)s * T_ + t0 + tok] = (float)besti;
    }
    __syncthreads();

    // residual update (f32 master) + bf16 mirror + commit = |new residual|^2
    const float* cbf = cb_f32 + (size_t)s * K_ * D_;
    for (int i = 0; i < (TT * D_) / 256; ++i) {
      const int e = i * 256 + tid;
      const int t = e >> 7;
      const int d = e & (D_ - 1);
      const float q  = cbf[(size_t)winners[t] * D_ + d];
      const float nr = Rf[e] - q;
      Rf[e]  = nr;
      Rtb[e] = f2bf(nr);
      commitAcc += nr * nr;
    }
    __syncthreads();
  }

  // ---- Stage 3: q_sum = proj_init - residual_final; out = w_out @ q_sum + b_out
  for (int i = 0; i < (TT * D_) / 256; ++i) {
    const int e = i * 256 + tid;
    Rtb[e] = f2bf(Pj[e] - Rf[e]);
  }
  __syncthreads();

  v16bf qfrag[4];
#pragma unroll
  for (int dsp = 0; dsp < 4; ++dsp) {
    const uint4* bp = (const uint4*)(Rtb + tok * D_ + dsp * 32 + hf * 16);
    qfrag[dsp] = pack_frag(bp[0], bp[1]);
  }
  for (int mt = 0; mt < DIN / 16; ++mt) {
    v8f acco = vzero;
    __builtin_prefetch(w_out_h + (size_t)((mt + 1) * 16 + lrow) * D_, 0, 1);
#pragma unroll
    for (int dsp = 0; dsp < 4; ++dsp) {
      const unsigned short* ap =
          w_out_h + (size_t)(mt * 16 + lrow) * D_ + dsp * 32 + hf * 16;
      const v16bf af = pack_frag(*(const uint4*)ap, *(const uint4*)(ap + 8));
      acco = __builtin_amdgcn_wmma_f32_16x16x32_bf16(
          false, af, false, qfrag[dsp], (short)0, acco, false, false);
    }
#pragma unroll
    for (int r = 0; r < 8; ++r) {
      const int m = mt * 16 + hf * 8 + r;
      out_y[(size_t)b * DIN * T_ + (size_t)m * T_ + t0 + tok] = acco[r] + b_out[m];
    }
  }

  // ---- commit loss block reduction -> atomic add
  red[tid] = commitAcc;
  __syncthreads();
  for (int off = 128; off > 0; off >>= 1) {
    if (tid < off) red[tid] += red[tid + off];
    __syncthreads();
  }
  if (tid == 0)
    atomicAdd(out_commit,
              red[0] * (1.0f / ((float)NCB * (float)B_ * (float)T_ * (float)D_)));
}

// ---------------- host launcher --------------------------------------------
extern "C" void kernel_launch(void* const* d_in, const int* in_sizes, int n_in,
                              void* d_out, int out_size, void* d_ws, size_t ws_size,
                              hipStream_t stream) {
  const float* z      = (const float*)d_in[0];
  const float* w_in   = (const float*)d_in[1];
  const float* b_in   = (const float*)d_in[2];
  const float* w_out  = (const float*)d_in[3];
  const float* b_out  = (const float*)d_in[4];
  const float* cbooks = (const float*)d_in[5];

  // workspace layout (bf16 copies + half-norms), ~4.5 MB
  char* ws = (char*)d_ws;
  unsigned short* cbh = (unsigned short*)ws;  ws += (size_t)NCB * K_ * D_ * 2;
  float*          cn  = (float*)ws;           ws += (size_t)NCB * K_ * 4;
  unsigned short* wih = (unsigned short*)ws;  ws += (size_t)D_ * DIN * 2;
  unsigned short* woh = (unsigned short*)ws;  ws += (size_t)DIN * D_ * 2;

  float* out_codes  = (float*)d_out;                         // [B][n][T]
  float* out_y      = out_codes + (size_t)B_ * NCB * T_;     // [B][Din][T]
  float* out_commit = out_y + (size_t)B_ * DIN * T_;         // scalar

  rvq_prep_cb<<<NCB * K_, 128, 0, stream>>>(cbooks, cbh, cn);
  rvq_prep_w<<<(2 * D_ * DIN + 255) / 256, 256, 0, stream>>>(w_in, w_out, wih, woh);
  rvq_zero_commit<<<1, 32, 0, stream>>>(out_commit);

  rvq_fused<<<dim3(T_ / TT, B_), 256, (size_t)SMEM_TOT, stream>>>(
      z, wih, b_in, woh, b_out, cbooks, cbh, cn, out_codes, out_y, out_commit);
}
